// PCPRender_16673063043683
// MI455X (gfx1250) — matrix operand order repaired and checked by reading, not compiled, not run on hardware
//
#include <hip/hip_runtime.h>
#include <hip/hip_bf16.h>

// ---------------------------------------------------------------------------
// PCP render: point splatting + small U-Net, WMMA implicit-GEMM convs (gfx1250)
// ---------------------------------------------------------------------------

#define HH 512
#define WW 512
#define BB 2
#define NN 262144
#define FDIM 32
#define HW (HH*WW)

typedef __attribute__((ext_vector_type(16))) _Float16 v16h;
typedef __attribute__((ext_vector_type(8)))  _Float16 v8h;
typedef __attribute__((ext_vector_type(8)))  float    v8f;
typedef __attribute__((ext_vector_type(4)))  unsigned int v4u;

// ---------------------------------------------------------------------------
// Camera prep: Einv (3x4) and Kinv (3x3) per batch
// ---------------------------------------------------------------------------
__device__ inline void inv3x3(const float m[9], float o[9]) {
  float a=m[0],b=m[1],c=m[2],d=m[3],e=m[4],f=m[5],g=m[6],h=m[7],i=m[8];
  float A =  (e*i - f*h);
  float Bc = -(d*i - f*g);
  float C =  (d*h - e*g);
  float det = a*A + b*Bc + c*C;
  float r = 1.0f / det;
  o[0]=A*r;  o[1]=-(b*i-c*h)*r; o[2]= (b*f-c*e)*r;
  o[3]=Bc*r; o[4]= (a*i-c*g)*r; o[5]=-(a*f-c*d)*r;
  o[6]=C*r;  o[7]=-(a*h-b*g)*r; o[8]= (a*e-b*d)*r;
}

__global__ void prep_cam_kernel(const float* __restrict__ K,
                                const float* __restrict__ E,
                                float* __restrict__ Einv,   // B*12 (3x4 row major)
                                float* __restrict__ Kinv) { // B*9
  int b = threadIdx.x;
  if (b >= BB) return;
  float R[9], Ri[9];
  const float* Eb = E + b*16;
  for (int r = 0; r < 3; ++r)
    for (int c = 0; c < 3; ++c) R[r*3+c] = Eb[r*4+c];
  inv3x3(R, Ri);
  float tx = Eb[3], ty = Eb[7], tz = Eb[11];
  for (int r = 0; r < 3; ++r) {
    Einv[b*12 + r*4 + 0] = Ri[r*3+0];
    Einv[b*12 + r*4 + 1] = Ri[r*3+1];
    Einv[b*12 + r*4 + 2] = Ri[r*3+2];
    Einv[b*12 + r*4 + 3] = -(Ri[r*3+0]*tx + Ri[r*3+1]*ty + Ri[r*3+2]*tz);
  }
  float Km[9], Ki[9];
  for (int j = 0; j < 9; ++j) Km[j] = K[b*9+j];
  inv3x3(Km, Ki);
  for (int j = 0; j < 9; ++j) Kinv[b*9+j] = Ki[j];
}

// ---------------------------------------------------------------------------
// Splat buffers init
// ---------------------------------------------------------------------------
__global__ void init_bufs_kernel(unsigned int* __restrict__ depth,
                                 int* __restrict__ idx) {
  int gid = blockIdx.x * blockDim.x + threadIdx.x;
  int total = BB * (HW + 1);
  if (gid < total) {
    depth[gid] = 0x7F800000u;  // +inf bits
    idx[gid]   = NN;
  }
}

// ---------------------------------------------------------------------------
// Project points + depth atomicMin (positive floats: uint ordering == float)
// ---------------------------------------------------------------------------
__global__ void project_kernel(const float* __restrict__ pts,
                               const float* __restrict__ K,
                               const float* __restrict__ Einv,
                               const float* __restrict__ nf,
                               unsigned int* __restrict__ depth,
                               int* __restrict__ pix_out,
                               float* __restrict__ z_out) {
  int gid = blockIdx.x * blockDim.x + threadIdx.x;
  if (gid >= BB * NN) return;
  int b = gid / NN;
  const float* p  = pts + (size_t)gid * 3;
  const float* Ei = Einv + b*12;
  const float* Kb = K + b*9;
  const float* nb = nf + b*3;
  float px = p[0], py = p[1], pz = p[2];
  float cx = Ei[0]*px + Ei[1]*py + Ei[2]*pz  + Ei[3];
  float cy = Ei[4]*px + Ei[5]*py + Ei[6]*pz  + Ei[7];
  float cz = Ei[8]*px + Ei[9]*py + Ei[10]*pz + Ei[11];
  float u0 = Kb[0]*cx + Kb[1]*cy + Kb[2]*cz;
  float v0 = Kb[3]*cx + Kb[4]*cy + Kb[5]*cz;
  float z  = Kb[6]*cx + Kb[7]*cy + Kb[8]*cz;
  float zs = (fabsf(z) > 1e-8f) ? z : 1e-8f;
  float uf = floorf(u0 / zs);
  float vf = floorf(v0 / zs);
  bool valid = (z > nb[0]) && (z < nb[1]) &&
               (uf >= 0.0f) && (uf < (float)WW) &&
               (vf >= 0.0f) && (vf < (float)HH);
  int pix = HW;
  if (valid) {
    int ui = (int)uf, vi = (int)vf;
    pix = vi * WW + ui;
    atomicMin(depth + (size_t)b*(HW+1) + pix, __float_as_uint(z));
  }
  pix_out[gid] = pix;
  z_out[gid]   = z;
}

// ---------------------------------------------------------------------------
// Winner pass: min point index among those matching the z-buffer
// ---------------------------------------------------------------------------
__global__ void winner_kernel(const unsigned int* __restrict__ depth,
                              const int* __restrict__ pix_in,
                              const float* __restrict__ z_in,
                              int* __restrict__ idx) {
  int gid = blockIdx.x * blockDim.x + threadIdx.x;
  if (gid >= BB * NN) return;
  int b = gid / NN;
  int i = gid % NN;
  int pix = pix_in[gid];
  if (pix >= HW) return;
  float z = z_in[gid];
  float dmin = __uint_as_float(depth[(size_t)b*(HW+1) + pix]);
  if (z <= dmin) atomicMin(idx + (size_t)b*(HW+1) + pix, i);
}

// ---------------------------------------------------------------------------
// Gather: out_feature / out_depth / ray dirs d + packed f16 NHWC conv input
// (64 channels: 0..31 feats, 32..34 dirs, 35..63 zero)
// ---------------------------------------------------------------------------
__global__ void gather_kernel(const int* __restrict__ idx,
                              const float* __restrict__ z_pts,
                              const float* __restrict__ feats,
                              const float* __restrict__ dflt,
                              const float* __restrict__ Kinv,
                              const float* __restrict__ E,
                              float* __restrict__ out_depth,   // (B,1,H,W)
                              float* __restrict__ out_feat,    // (B,32,H,W)
                              float* __restrict__ out_d,       // (B,3,H,W)
                              _Float16* __restrict__ fbuf) {   // (B,H,W,64)
  int gid = blockIdx.x * blockDim.x + threadIdx.x;
  if (gid >= BB * HW) return;
  int b = gid / HW;
  int p = gid % HW;
  int y = p / WW, x = p % WW;

  int i = idx[(size_t)b*(HW+1) + p];
  float depth = (i < NN) ? z_pts[(size_t)b*NN + i] : 0.0f;
  out_depth[(size_t)b*HW + p] = depth;

  const float* fsrc = (i < NN) ? (feats + ((size_t)b*NN + i)*FDIM)
                               : (dflt + (size_t)b*FDIM);
  _Float16* fd = fbuf + ((size_t)b*HW + p)*64;
  for (int c = 0; c < FDIM; ++c) {
    float v = fsrc[c];
    out_feat[((size_t)b*FDIM + c)*HW + p] = v;
    fd[c] = (_Float16)v;
  }

  // ray dir: dc = Kinv*(x,y,1); dw = E[:3,:3]*dc + E[:3,3]; normalize
  const float* Ki = Kinv + b*9;
  const float* Eb = E + b*16;
  float fx = (float)x, fy = (float)y;
  float dcx = Ki[0]*fx + Ki[1]*fy + Ki[2];
  float dcy = Ki[3]*fx + Ki[4]*fy + Ki[5];
  float dcz = Ki[6]*fx + Ki[7]*fy + Ki[8];
  float dwx = Eb[0]*dcx + Eb[1]*dcy + Eb[2]*dcz  + Eb[3];
  float dwy = Eb[4]*dcx + Eb[5]*dcy + Eb[6]*dcz  + Eb[7];
  float dwz = Eb[8]*dcx + Eb[9]*dcy + Eb[10]*dcz + Eb[11];
  float nrm = sqrtf(dwx*dwx + dwy*dwy + dwz*dwz);
  float s = 1.0f / fmaxf(nrm, 1e-12f);
  float dx = dwx*s, dy = dwy*s, dz = dwz*s;
  if (depth == 0.0f) { dx = 0.0f; dy = 0.0f; dz = 0.0f; }
  out_d[((size_t)b*3 + 0)*HW + p] = dx;
  out_d[((size_t)b*3 + 1)*HW + p] = dy;
  out_d[((size_t)b*3 + 2)*HW + p] = dz;
  fd[FDIM + 0] = (_Float16)dx;
  fd[FDIM + 1] = (_Float16)dy;
  fd[FDIM + 2] = (_Float16)dz;
  for (int c = FDIM + 3; c < 64; ++c) fd[c] = (_Float16)0.0f;
}

// ---------------------------------------------------------------------------
// Weight repack: OIHW f32 -> f16 [tap(9)][chunk(CinPad/32)][n(Cout)][c(32)]
// so each lane's WMMA B fragment is two 16B runs (c 8h..8h+7 and 16+8h..16+8h+7).
// ---------------------------------------------------------------------------
__global__ void pack_weights_kernel(const float* __restrict__ w,
                                    _Float16* __restrict__ wp,
                                    int Cout, int CinReal, int CinPad) {
  int nch = CinPad / 32;
  int total = 9 * nch * Cout * 32;
  int gid = blockIdx.x * blockDim.x + threadIdx.x;
  if (gid >= total) return;
  int c   = gid % 32;
  int n   = (gid / 32) % Cout;
  int ch  = (gid / (32 * Cout)) % nch;
  int tap = gid / (32 * Cout * nch);
  int cin = ch * 32 + c;
  float v = 0.0f;
  if (cin < CinReal) {
    int ky = tap / 3, kx = tap % 3;
    v = w[(((size_t)n * CinReal + cin) * 3 + ky) * 3 + kx];
  }
  wp[gid] = (_Float16)v;
}

// ---------------------------------------------------------------------------
// 3x3 SAME conv, implicit GEMM on WMMA f16->f32.
// 256 threads = 8 waves. Each wave owns one 16-channel output slice (ct) and
// FOUR pixel tiles at once (4 x v8f accumulators): each weight (B) fragment is
// loaded once per K-step and feeds 4 WMMAs. The 4 A fragments are loaded into
// DISTINCT live values before the 4 WMMAs so the ds_load_b128s pipeline as one
// clause instead of serializing on a recycled register (one dscnt wait per
// K-step instead of four). COUT=64 -> 4 slices x 2 pixel halves (block tile
// 128 px); COUT=128 -> 8 slices (block tile 64 px).
// LDS: 3 rows x (TW+2) cols x CIN f16 input tile.
// CONCAT: channels 0..127 from half-res x2 (nearest up2), 128..191 from x1.
// ---------------------------------------------------------------------------
template<int CIN, int COUT, bool RELU, bool CONCAT>
__global__ __launch_bounds__(256)
void conv3x3_wmma_kernel(const _Float16* __restrict__ in,
                         const _Float16* __restrict__ in2,
                         const _Float16* __restrict__ wpack,
                         const float* __restrict__ bias,
                         _Float16* __restrict__ out,
                         int Wd, int Hd) {
  constexpr int NCH   = CIN / 32;
  constexpr int NTILE = COUT / 16;       // output-channel slices
  constexpr int PHG   = 8 / NTILE;       // pixel halves covered by the block
  constexpr int TW    = 64 * PHG;        // pixels per block (row segment)
  static_assert(PHG >= 1 && PHG * NTILE == 8, "wave partition");

  extern __shared__ char smem_raw[];
  _Float16* s = (_Float16*)smem_raw;     // [3][TW+2][CIN]

  const int x0  = blockIdx.x * TW;
  const int y   = blockIdx.y;
  const int b   = blockIdx.z;
  const int tid = threadIdx.x;

  // ---- stage input tile (with halo, zero-padded) into LDS ----
  constexpr int CHUNKS = CIN / 8;        // 16B chunks per pixel
  for (int idx = tid; idx < 3 * (TW + 2) * CHUNKS; idx += 256) {
    int cc  = idx % CHUNKS;
    int t   = idx / CHUNKS;
    int col = t % (TW + 2);
    int r   = t / (TW + 2);
    int gy = y - 1 + r;
    int gx = x0 - 1 + col;
    v4u val = {0u, 0u, 0u, 0u};
    if (gy >= 0 && gy < Hd && gx >= 0 && gx < Wd) {
      if (!CONCAT) {
        val = *(const v4u*)(in + ((size_t)b * Hd * Wd + (size_t)gy * Wd + gx) * CIN + cc * 8);
      } else {
        int c8 = cc * 8;
        if (c8 < 128) {
          int W2 = Wd >> 1, H2 = Hd >> 1;
          val = *(const v4u*)(in + ((size_t)b * H2 * W2 + (size_t)(gy >> 1) * W2 + (gx >> 1)) * 128 + c8);
        } else {
          val = *(const v4u*)(in2 + ((size_t)b * Hd * Wd + (size_t)gy * Wd + gx) * 64 + (c8 - 128));
        }
      }
    }
    *(v4u*)(s + (size_t)(r * (TW + 2) + col) * CIN + cc * 8) = val;
  }
  __syncthreads();

  const int lane   = tid & 31;
  const int wave   = tid >> 5;
  const int halfHi = (lane >= 16) ? 1 : 0;
  const int l16    = lane & 15;

  const int ct = wave % NTILE;           // output-channel slice
  const int ph = wave / NTILE;           // 64-pixel half within block tile
  const int n0 = ct * 16;
  const int p0 = ph * 64;                // base pixel (local)

  v8f acc0 = {}, acc1 = {}, acc2 = {}, acc3 = {};

  for (int tap = 0; tap < 9; ++tap) {
    int dy = tap / 3, dx = tap % 3;
    const _Float16* arow = s + (size_t)(dy * (TW + 2)) * CIN;
    #pragma unroll
    for (int ch = 0; ch < NCH; ++ch) {
      // B fragment (weights): lane column n0+l16, K 8h..8h+7 / 16+8h..16+8h+7 -- once.
      const _Float16* bp = wpack +
          ((size_t)(tap * NCH + ch) * COUT + (n0 + l16)) * 32 + (halfHi ? 8 : 0);
      v8h blo = *(const v8h*)bp;
      v8h bhi = *(const v8h*)(bp + 16);
      v16h bm;
      #pragma unroll
      for (int e = 0; e < 8; ++e) { bm[e] = blo[e]; bm[e + 8] = bhi[e]; }

      int k0 = ch * 32 + (halfHi ? 8 : 0);
      const _Float16* abase = arow + (size_t)(p0 + l16 + dx) * CIN + k0;

      // Load all four A fragments first (distinct live values -> one clause).
      v16h a0, a1, a2, a3;
      {
        const _Float16* ap = abase;
        v8h lo = *(const v8h*)ap, hi = *(const v8h*)(ap + 16);
        #pragma unroll
        for (int e = 0; e < 8; ++e) { a0[e] = lo[e]; a0[e + 8] = hi[e]; }
      }
      {
        const _Float16* ap = abase + (size_t)16 * CIN;
        v8h lo = *(const v8h*)ap, hi = *(const v8h*)(ap + 16);
        #pragma unroll
        for (int e = 0; e < 8; ++e) { a1[e] = lo[e]; a1[e + 8] = hi[e]; }
      }
      {
        const _Float16* ap = abase + (size_t)32 * CIN;
        v8h lo = *(const v8h*)ap, hi = *(const v8h*)(ap + 16);
        #pragma unroll
        for (int e = 0; e < 8; ++e) { a2[e] = lo[e]; a2[e + 8] = hi[e]; }
      }
      {
        const _Float16* ap = abase + (size_t)48 * CIN;
        v8h lo = *(const v8h*)ap, hi = *(const v8h*)(ap + 16);
        #pragma unroll
        for (int e = 0; e < 8; ++e) { a3[e] = lo[e]; a3[e + 8] = hi[e]; }
      }

      acc0 = __builtin_amdgcn_wmma_f32_16x16x32_f16(false, a0, false, bm, (short)0, acc0, false, false);
      acc1 = __builtin_amdgcn_wmma_f32_16x16x32_f16(false, a1, false, bm, (short)0, acc1, false, false);
      acc2 = __builtin_amdgcn_wmma_f32_16x16x32_f16(false, a2, false, bm, (short)0, acc2, false, false);
      acc3 = __builtin_amdgcn_wmma_f32_16x16x32_f16(false, a3, false, bm, (short)0, acc3, false, false);
    }
  }

  // ---- store: fused bias + relu; C/D layout: elem r -> M=r+8*halfHi, N=l16
  float bn = bias[n0 + l16];
  const size_t outBase = (size_t)b * Hd * Wd * COUT;
  #pragma unroll
  for (int q = 0; q < 4; ++q) {
    const v8f& acc = (q == 0) ? acc0 : (q == 1) ? acc1 : (q == 2) ? acc2 : acc3;
    int mbase = x0 + p0 + q * 16 + (halfHi ? 8 : 0);
    #pragma unroll
    for (int r2 = 0; r2 < 8; ++r2) {
      float v = acc[r2] + bn;
      if (RELU) v = (v > 0.0f) ? v : 0.0f;
      out[outBase + ((size_t)y * Wd + (mbase + r2)) * COUT + (n0 + l16)] = (_Float16)v;
    }
  }
}

// ---------------------------------------------------------------------------
// 2x2 maxpool, NHWC f16, 512^2x64 -> 256^2x64
// ---------------------------------------------------------------------------
__global__ void maxpool2_kernel(const _Float16* __restrict__ in,
                                _Float16* __restrict__ out) {
  const int H2 = HH / 2, W2 = WW / 2, C = 64, CC = C / 8;
  int gid = blockIdx.x * blockDim.x + threadIdx.x;
  int total = BB * H2 * W2 * CC;
  if (gid >= total) return;
  int cc = gid % CC;
  int t  = gid / CC;
  int x2 = t % W2;
  int y2 = (t / W2) % H2;
  int b  = t / (W2 * H2);
  size_t base = ((size_t)b * HH * WW + (size_t)(2 * y2) * WW + 2 * x2) * C + cc * 8;
  v8h a0 = *(const v8h*)(in + base);
  v8h a1 = *(const v8h*)(in + base + C);
  v8h a2 = *(const v8h*)(in + base + (size_t)WW * C);
  v8h a3 = *(const v8h*)(in + base + (size_t)WW * C + C);
  v8h m;
  #pragma unroll
  for (int e = 0; e < 8; ++e) {
    _Float16 v = a0[e];
    v = (a1[e] > v) ? a1[e] : v;
    v = (a2[e] > v) ? a2[e] : v;
    v = (a3[e] > v) ? a3[e] : v;
    m[e] = v;
  }
  *(v8h*)(out + ((size_t)b * H2 * W2 + (size_t)y2 * W2 + x2) * C + cc * 8) = m;
}

// ---------------------------------------------------------------------------
// 1x1 conv 64->3, f32 NCHW output (final 'x')
// ---------------------------------------------------------------------------
__global__ void conv1x1_out_kernel(const _Float16* __restrict__ in,
                                   const float* __restrict__ w,   // (3,64)
                                   const float* __restrict__ bias,
                                   float* __restrict__ out) {
  int gid = blockIdx.x * blockDim.x + threadIdx.x;
  if (gid >= BB * HW) return;
  int b = gid / HW;
  int p = gid % HW;
  const _Float16* src = in + ((size_t)b * HW + p) * 64;
  float a0 = bias[0], a1 = bias[1], a2 = bias[2];
  #pragma unroll 8
  for (int c = 0; c < 64; ++c) {
    float v = (float)src[c];
    a0 += v * w[c];
    a1 += v * w[64 + c];
    a2 += v * w[128 + c];
  }
  out[((size_t)b * 3 + 0) * HW + p] = a0;
  out[((size_t)b * 3 + 1) * HW + p] = a1;
  out[((size_t)b * 3 + 2) * HW + p] = a2;
}

// ---------------------------------------------------------------------------
// Host-side launch
// ---------------------------------------------------------------------------
static inline size_t alignup(size_t v) { return (v + 255) & ~(size_t)255; }

extern "C" void kernel_launch(void* const* d_in, const int* in_sizes, int n_in,
                              void* d_out, int out_size, void* d_ws, size_t ws_size,
                              hipStream_t stream) {
  (void)in_sizes; (void)n_in; (void)out_size; (void)ws_size;

  const float* point_features = (const float*)d_in[0];
  const float* default_feats  = (const float*)d_in[1];
  const float* point_clouds   = (const float*)d_in[2];
  const float* cam_K          = (const float*)d_in[3];
  const float* cam_E          = (const float*)d_in[4];
  const float* near_far       = (const float*)d_in[5];
  const float* w1 = (const float*)d_in[7];  const float* b1 = (const float*)d_in[8];
  const float* w2 = (const float*)d_in[9];  const float* b2 = (const float*)d_in[10];
  const float* w3 = (const float*)d_in[11]; const float* b3 = (const float*)d_in[12];
  const float* w4 = (const float*)d_in[13]; const float* b4 = (const float*)d_in[14];
  const float* w5 = (const float*)d_in[15]; const float* b5 = (const float*)d_in[16];
  const float* w6 = (const float*)d_in[17]; const float* b6 = (const float*)d_in[18];
  const float* w7 = (const float*)d_in[19]; const float* b7 = (const float*)d_in[20];

  // d_out layout: x(B,3,H,W) | depth(B,1,H,W) | feat(B,32,H,W) | d(B,3,H,W)
  float* out_x     = (float*)d_out;
  float* out_depth = out_x + (size_t)BB * 3 * HW;
  float* out_feat  = out_depth + (size_t)BB * 1 * HW;
  float* out_d     = out_feat + (size_t)BB * FDIM * HW;

  // ---- workspace carve-up ----
  char* ws = (char*)d_ws;
  size_t o = 0;
  auto take = [&](size_t bytes) { char* p = ws + o; o = alignup(o + bytes); return p; };

  float*        camEinv = (float*)take(BB * 12 * sizeof(float));
  float*        camKinv = (float*)take(BB * 9 * sizeof(float));
  unsigned int* depth   = (unsigned int*)take((size_t)BB * (HW + 1) * 4);
  int*          idxb    = (int*)take((size_t)BB * (HW + 1) * 4);
  int*          pix     = (int*)take((size_t)BB * NN * 4);
  float*        zpts    = (float*)take((size_t)BB * NN * 4);
  _Float16* wp1 = (_Float16*)take((size_t)9 * 2 * 64  * 32 * 2);
  _Float16* wp2 = (_Float16*)take((size_t)9 * 2 * 64  * 32 * 2);
  _Float16* wp3 = (_Float16*)take((size_t)9 * 2 * 128 * 32 * 2);
  _Float16* wp4 = (_Float16*)take((size_t)9 * 4 * 128 * 32 * 2);
  _Float16* wp5 = (_Float16*)take((size_t)9 * 6 * 64  * 32 * 2);
  _Float16* wp6 = (_Float16*)take((size_t)9 * 2 * 64  * 32 * 2);
  _Float16* fbuf = (_Float16*)take((size_t)BB * HW * 64 * 2);          // conv input; reused as x3a
  _Float16* t1   = (_Float16*)take((size_t)BB * HW * 64 * 2);          // reused as x3b
  _Float16* x1   = (_Float16*)take((size_t)BB * HW * 64 * 2);
  _Float16* pbuf = (_Float16*)take((size_t)BB * (HW / 4) * 64 * 2);
  _Float16* t2   = (_Float16*)take((size_t)BB * (HW / 4) * 128 * 2);
  _Float16* x2   = (_Float16*)take((size_t)BB * (HW / 4) * 128 * 2);

  // ---- stage 1: splat ----
  prep_cam_kernel<<<1, 32, 0, stream>>>(cam_K, cam_E, camEinv, camKinv);
  {
    int total = BB * (HW + 1);
    init_bufs_kernel<<<(total + 255) / 256, 256, 0, stream>>>(depth, idxb);
  }
  {
    int total = BB * NN;
    project_kernel<<<(total + 255) / 256, 256, 0, stream>>>(
        point_clouds, cam_K, camEinv, near_far, depth, pix, zpts);
    winner_kernel<<<(total + 255) / 256, 256, 0, stream>>>(depth, pix, zpts, idxb);
  }
  {
    int total = BB * HW;
    gather_kernel<<<(total + 255) / 256, 256, 0, stream>>>(
        idxb, zpts, point_features, default_feats, camKinv, cam_E,
        out_depth, out_feat, out_d, fbuf);
  }

  // ---- weight repack (f16, WMMA-B layout) ----
  auto pack = [&](const float* w, _Float16* wp, int Cout, int CinReal, int CinPad) {
    int total = 9 * (CinPad / 32) * Cout * 32;
    pack_weights_kernel<<<(total + 255) / 256, 256, 0, stream>>>(w, wp, Cout, CinReal, CinPad);
  };
  pack(w1, wp1, 64, 35, 64);
  pack(w2, wp2, 64, 64, 64);
  pack(w3, wp3, 128, 64, 64);
  pack(w4, wp4, 128, 128, 128);
  pack(w5, wp5, 64, 192, 192);
  pack(w6, wp6, 64, 64, 64);

  // ---- stage 2: U-Net via WMMA implicit GEMM ----
  // COUT=64 -> block tile 128 px; COUT=128 -> 64 px.
  const size_t lds_c64_t128  = (size_t)3 * 130 * 64  * 2;  // conv1/2/6
  const size_t lds_c64_t64   = (size_t)3 * 66  * 64  * 2;  // conv3
  const size_t lds_c128_t64  = (size_t)3 * 66  * 128 * 2;  // conv4
  const size_t lds_c192_t128 = (size_t)3 * 130 * 192 * 2;  // conv5

  dim3 g512_t128(WW / 128, HH, BB);
  dim3 g256_t64((WW / 2) / 64, HH / 2, BB);

  conv3x3_wmma_kernel<64, 64, true, false><<<g512_t128, 256, lds_c64_t128, stream>>>(
      fbuf, nullptr, wp1, b1, t1, WW, HH);
  conv3x3_wmma_kernel<64, 64, true, false><<<g512_t128, 256, lds_c64_t128, stream>>>(
      t1, nullptr, wp2, b2, x1, WW, HH);
  {
    int total = BB * (HH / 2) * (WW / 2) * (64 / 8);
    maxpool2_kernel<<<(total + 255) / 256, 256, 0, stream>>>(x1, pbuf);
  }
  conv3x3_wmma_kernel<64, 128, true, false><<<g256_t64, 256, lds_c64_t64, stream>>>(
      pbuf, nullptr, wp3, b3, t2, WW / 2, HH / 2);
  conv3x3_wmma_kernel<128, 128, true, false><<<g256_t64, 256, lds_c128_t64, stream>>>(
      t2, nullptr, wp4, b4, x2, WW / 2, HH / 2);
  // conv5: implicit concat(up2(x2), x1) -> 192 ch; writes into fbuf (x3a)
  conv3x3_wmma_kernel<192, 64, true, true><<<g512_t128, 256, lds_c192_t128, stream>>>(
      x2, x1, wp5, b5, fbuf, WW, HH);
  conv3x3_wmma_kernel<64, 64, true, false><<<g512_t128, 256, lds_c64_t128, stream>>>(
      fbuf, nullptr, wp6, b6, t1, WW, HH);
  {
    int total = BB * HW;
    conv1x1_out_kernel<<<(total + 255) / 256, 256, 0, stream>>>(t1, w7, b7, out_x);
  }
}
